// Maxsim_63239098466666
// MI455X (gfx1250) — matrix operand order repaired and checked by reading, not compiled, not run on hardware
//
#include <hip/hip_runtime.h>

typedef __attribute__((ext_vector_type(16))) _Float16 v16h;
typedef __attribute__((ext_vector_type(8)))  _Float16 v8h;
typedef __attribute__((ext_vector_type(8)))  float    v8f;

#define Q_SIZE       16
#define QUERY_MAXLEN 32
#define DIM          128
#define D_SIZE       2048
#define DOC_MAXLEN   180
#define STILES       12            // ceil(180/16); tiles 0..10 full, tile 11 partial
#define PITCH        (DIM + 8)     // f16 pitch; +16B to break LDS bank conflicts

// Raw v_max_num_f32: skips LLVM's sNaN canonicalization that triples the
// VALU cost of libm fmaxf. WMMA output is never sNaN, so this is exact.
__device__ __forceinline__ float vmax_f32(float a, float b) {
    float d;
    asm("v_max_num_f32 %0, %1, %2" : "=v"(d) : "v"(a), "v"(b));
    return d;
}

__global__ __launch_bounds__(512, 1)
void maxsim_wmma_kernel(const float* __restrict__ Qg,
                        const float* __restrict__ Dg,
                        float* __restrict__ out)
{
    // double-buffered f16 doc tile: 2 x 16 tokens x (128+8) halfs = 8.5 KB
    __shared__ _Float16 tile[2][16 * PITCH];

    const int k    = blockIdx.x;           // document
    const int lane = threadIdx.x & 31;     // wave32
    const int q    = threadIdx.x >> 5;     // wave id == query id (16 waves)

    // ---------------- A fragments: Q[q] (32x128) f32 -> f16 ----------------
    // CDNA5 16-bit A layout: lanes 0-15 -> M=lane,    K = {0..7, 16..23} (+kt*32)
    //                        lanes16-31 -> M=lane-16, K = {8..15, 24..31}
    const int mrow = lane & 15;
    const int akb  = (lane < 16) ? 0 : 8;
    v16h afrag[2][4];
#pragma unroll
    for (int mt = 0; mt < 2; ++mt) {
        const float* qrow = Qg + (size_t)(q * QUERY_MAXLEN + mt * 16 + mrow) * DIM;
#pragma unroll
        for (int kt = 0; kt < 4; ++kt) {
            float tmp[16];
            const float4* p1 = (const float4*)(qrow + kt * 32 + akb);
            const float4* p2 = (const float4*)(qrow + kt * 32 + akb + 16);
            *(float4*)&tmp[0]  = p1[0];
            *(float4*)&tmp[4]  = p1[1];
            *(float4*)&tmp[8]  = p2[0];
            *(float4*)&tmp[12] = p2[1];
            v16h a;
#pragma unroll
            for (int e = 0; e < 16; ++e) a[e] = (_Float16)tmp[e];
            afrag[mt][kt] = a;
        }
    }

    // ------------- software-pipelined stage: 16x128 f32 -> f16 LDS ---------
    const float* dbase = Dg + (size_t)k * DOC_MAXLEN * DIM;
    const int eidx = threadIdx.x * 4;    // 512 threads * 4 elems = 2048 = 16*128
    const int ls   = eidx >> 7;          // local token 0..15
    const int lcol = eidx & 127;         // dim column

    // Phase 1: issue the global load (no consumer nearby -> latency hidden
    // under the WMMA chain issued between phase 1 and phase 2).
    auto gload = [&](int st) -> float4 {
        const int srow = st * 16 + ls;
        return *(const float4*)(dbase + (size_t)srow * DIM + lcol);
    };
    auto gload_tail = [&]() -> float4 {   // tile 11: tokens 176..191
        const int srow = (STILES - 1) * 16 + ls;
        float4 v = make_float4(0.f, 0.f, 0.f, 0.f);   // zeros keep WMMA NaN-free
        if (srow < DOC_MAXLEN)
            v = *(const float4*)(dbase + (size_t)srow * DIM + lcol);
        return v;
    };
    // Phase 2: convert (v_cvt_pk_f16_f32) and store to LDS.
    auto cvtstore = [&](float4 v, int buf) {
        _Float16* dst = &tile[buf][ls * PITCH + lcol];
        dst[0] = (_Float16)v.x;
        dst[1] = (_Float16)v.y;
        dst[2] = (_Float16)v.z;
        dst[3] = (_Float16)v.w;
    };

    // CDNA5 16-bit B layout: lanes 0-15 -> N=lane,    K = {0..7, 16..23}  (+kt*32)
    //                        lanes16-31 -> N=lane-16, K = {8..15, 24..31}
    const int n   = lane & 15;
    const int bkb = (lane < 16) ? 0 : 8;

    // One 16x16 sim tile per m-tile: 8 WMMAs over K=128.
    auto sim_tile = [&](int buf, v8f& acc0, v8f& acc1) {
        const _Float16* brow = &tile[buf][n * PITCH + bkb];
        v16h bfrag[4];
#pragma unroll
        for (int kt = 0; kt < 4; ++kt) {
            v8h b0 = *(const v8h*)(brow + kt * 32);       // ds_load_b128
            v8h b1 = *(const v8h*)(brow + kt * 32 + 16);  // ds_load_b128
            v16h b;
#pragma unroll
            for (int e = 0; e < 8; ++e) { b[e] = b0[e]; b[e + 8] = b1[e]; }
            bfrag[kt] = b;
        }
#pragma unroll
        for (int kt = 0; kt < 4; ++kt) {
            acc0 = __builtin_amdgcn_wmma_f32_16x16x32_f16(false, afrag[0][kt], false,
                                                          bfrag[kt], (short)0, acc0,
                                                          false, false);
            acc1 = __builtin_amdgcn_wmma_f32_16x16x32_f16(false, afrag[1][kt], false,
                                                          bfrag[kt], (short)0, acc1,
                                                          false, false);
        }
    };

    v8f rmax0, rmax1;

    // ---- prologue: stage tile 0 ----
    cvtstore(gload(0), 0);
    __syncthreads();

    // ---- tile 0 (peeled): rmax initialized by copy, no max needed ----
    {
        float4 pend = gload(1);            // issue load for tile 1
        v8f acc0 = {}, acc1 = {};
        sim_tile(0, acc0, acc1);           // WMMAs cover the load latency
        cvtstore(pend, 1);                 // wait + cvt + ds_store after WMMAs
        rmax0 = acc0;
        rmax1 = acc1;
        __syncthreads();
    }

    // ---- tiles 1..9: all 16 doc tokens valid ----
    for (int st = 1; st < STILES - 2; ++st) {
        const int buf = st & 1;
        float4 pend = gload(st + 1);       // issue load for next tile
        v8f acc0 = {}, acc1 = {};
        sim_tile(buf, acc0, acc1);
        cvtstore(pend, buf ^ 1);
#pragma unroll
        for (int i = 0; i < 8; ++i) {
            rmax0[i] = vmax_f32(rmax0[i], acc0[i]);
            rmax1[i] = vmax_f32(rmax1[i], acc1[i]);
        }
        __syncthreads();                   // next iter overwrites buf we just read
    }

    // ---- tile 10: stage partial tile 11, compute tile 10 ----
    {
        float4 pend = gload_tail();        // tile 11 (zero-padded)
        v8f acc0 = {}, acc1 = {};
        sim_tile(0, acc0, acc1);           // tile 10 lives in buf 0
        cvtstore(pend, 1);
#pragma unroll
        for (int i = 0; i < 8; ++i) {
            rmax0[i] = vmax_f32(rmax0[i], acc0[i]);
            rmax1[i] = vmax_f32(rmax1[i], acc1[i]);
        }
        __syncthreads();
    }

    // ---- tail tile 11: tokens 176..191, only n < 4 valid ----
    {
        v8f acc0 = {}, acc1 = {};
        sim_tile(1, acc0, acc1);
        const float neg   = -__builtin_inff();
        const bool  valid = ((STILES - 1) * 16 + n) < DOC_MAXLEN;
#pragma unroll
        for (int i = 0; i < 8; ++i) {
            rmax0[i] = vmax_f32(rmax0[i], valid ? acc0[i] : neg);
            rmax1[i] = vmax_f32(rmax1[i], valid ? acc1[i] : neg);
        }
    }

    // ---------------- MaxSim reduction -------------------------------------
    // rmax[i]: lanes0-15 hold row M=i (N=lane), lanes16-31 hold row M=i+8.
    float sum = 0.f;
#pragma unroll
    for (int i = 0; i < 8; ++i) {
        float m0 = rmax0[i];
        float m1 = rmax1[i];
#pragma unroll
        for (int msk = 1; msk <= 8; msk <<= 1) {      // max over 16 N-lanes
            m0 = vmax_f32(m0, __shfl_xor(m0, msk, 32));
            m1 = vmax_f32(m1, __shfl_xor(m1, msk, 32));
        }
        sum += m0 + m1;                               // sum over m (8 rows/half)
    }
    sum += __shfl_xor(sum, 16, 32);                   // merge the two M-halves

    if (lane == 0) out[(size_t)q * D_SIZE + k] = sum;
}

extern "C" void kernel_launch(void* const* d_in, const int* in_sizes, int n_in,
                              void* d_out, int out_size, void* d_ws, size_t ws_size,
                              hipStream_t stream)
{
    (void)in_sizes; (void)n_in; (void)d_ws; (void)ws_size; (void)out_size;
    const float* Qg = (const float*)d_in[0];   // [16, 32, 128] f32
    const float* Dg = (const float*)d_in[1];   // [2048, 180, 128] f32
    float* out = (float*)d_out;                // [16, 2048] f32

    maxsim_wmma_kernel<<<dim3(D_SIZE), dim3(512), 0, stream>>>(Qg, Dg, out);
}
